// DirectionalModel_8426725834953
// MI455X (gfx1250) — matrix operand order repaired and checked by reading, not compiled, not run on hardware
//
#include <hip/hip_runtime.h>
#include <hip/hip_bf16.h>
#include <math.h>

#define BB 8
#define VV 2048
#define DD 16
#define FF 64
#define EMBD 32
#define FEAT 16
#define BIGF 1.0e9f
#define KW 4   // waves (=tiles) per block

typedef _Float16 h16;
typedef __attribute__((ext_vector_type(2)))  _Float16 v2h;
typedef __attribute__((ext_vector_type(8)))  _Float16 v8h;
typedef __attribute__((ext_vector_type(16))) _Float16 v16h;
typedef __attribute__((ext_vector_type(8)))  float    v8f;

static __device__ __forceinline__ v8f wmma32(v16h a, v16h b, v8f c) {
  return __builtin_amdgcn_wmma_f32_16x16x32_f16(false, a, false, b, (short)0, c, false, false);
}

// A fragment from f16 row-major LDS tile [16][64]; two contiguous 8-half runs
// per lane per ISA 16-bit A 16x32 layout -> 2x ds_load_b128.
static __device__ __forceinline__ v16h load_a_lds16(const h16* A, int k0, int lane) {
  const int m  = lane & 15;
  const int kb = (lane & 16) ? 8 : 0;
  const h16* p0 = A + m * 64 + k0 + kb;
  v16h a;
  ((v8h*)&a)[0] = *(const v8h*)p0;          // K = k0+kb .. +7
  ((v8h*)&a)[1] = *(const v8h*)(p0 + 16);   // K = k0+16+kb .. +7
  return a;
}

// B fragment pre-packed per lane: 32 contiguous bytes -> 2x global_load_b128.
static __device__ __forceinline__ v16h load_b_packed(const h16* P, int frag, int lane) {
  return *(const v16h*)(P + (((size_t)frag * 32 + lane) << 4));
}

static __device__ __forceinline__ float sigm(float x) { return 1.f / (1.f + expf(-x)); }

// ---------------- weight prep: pack f32 weights into per-lane WMMA-B fragments ----------------
// packed index t -> frag = t>>9, lane = (t>>4)&31, e = t&15
// source: W[k*N + n], k = kstep*32 + ((lane&16)?8:0) + (e&7) + ((e&8)?16:0), n = ntile*16 + (lane&15)
static __device__ __forceinline__ void pack_one(const float* W, h16* dst, int N, int t, int Kvalid) {
  int frag = t >> 9, r = t & 511;
  int lane = r >> 4, e = r & 15;
  int nt = N >> 4;
  int kstep = frag / nt, ntile = frag % nt;
  int k = kstep * 32 + ((lane & 16) ? 8 : 0) + (e & 7) + ((e & 8) ? 16 : 0);
  int n = ntile * 16 + (lane & 15);
  float v = (k < Kvalid) ? W[k * N + n] : 0.f;
  dst[t] = (h16)v;
}

__global__ void k_prep(const float* encW, const float* gatW, const float* gruWx,
                       const float* gruWh, const float* decW1,
                       h16* wenc, h16* wgat, h16* wgx, h16* wgh, h16* wd1) {
  int t = blockIdx.x * blockDim.x + threadIdx.x;
  if (t < 4096)  pack_one(encW,  wenc, 64,  t, 48);   // K padded 48->64
  if (t < 4096)  pack_one(gatW,  wgat, 64,  t, 64);
  if (t < 12288) { pack_one(gruWx, wgx, 192, t, 64);
                   pack_one(gruWh, wgh, 192, t, 64); }
  if (t < 2048)  pack_one(decW1, wd1, 32,  t, 64);
}

// ---------------- encoder: node_enc = [emb_c, feat] @ enc_W + enc_b ----------------
__global__ __launch_bounds__(32 * KW) void k_encode(const float* feat, const float* emb,
                                                    const h16* wenc, const float* encb,
                                                    float* node_enc) {
  __shared__ __align__(16) h16 A[KW][16 * 64];
  __shared__ float scale[KW][16];
  const int lane = threadIdx.x & 31, w = threadIdx.x >> 5;
  const int row0 = (blockIdx.x * KW + w) * 16;     // global row = b*V+v
  h16* Al = A[w];
  if (lane < 16) {
    int v = (row0 + lane) & (VV - 1);
    float s = 0.f;
    for (int j = 0; j < EMBD; ++j) { float e = emb[v * EMBD + j]; s += e * e; }
    scale[w][lane] = fminf(1.f, 1.f / fmaxf(sqrtf(s), 1e-12f));
  }
  __syncthreads();
  for (int idx = lane; idx < 16 * 32; idx += 32) {       // half-pairs
    int m = idx >> 5, jp = (idx & 31) << 1;
    int gr = row0 + m, v = gr & (VV - 1);
    float v0 = 0.f, v1 = 0.f;
    if (jp < EMBD)            { v0 = emb[v*EMBD + jp] * scale[w][m];
                                v1 = emb[v*EMBD + jp+1] * scale[w][m]; }
    else if (jp < EMBD+FEAT)  { v0 = feat[gr*FEAT + (jp-EMBD)];
                                v1 = feat[gr*FEAT + (jp-EMBD+1)]; }
    v2h p; p[0] = (h16)v0; p[1] = (h16)v1;
    ((v2h*)Al)[idx] = p;
  }
  __syncthreads();
  v16h a0 = load_a_lds16(Al, 0, lane);
  v16h a1 = load_a_lds16(Al, 32, lane);
  const int n = lane & 15, m0 = (lane & 16) ? 8 : 0;
  for (int j = 0; j < 4; ++j) {
    v8f acc = {};
    acc = wmma32(a0, load_b_packed(wenc, 0 + j, lane), acc);
    acc = wmma32(a1, load_b_packed(wenc, 4 + j, lane), acc);
    float bias = encb[j * 16 + n];
#pragma unroll
    for (int r = 0; r < 8; ++r)
      node_enc[(row0 + m0 + r) * FF + j * 16 + n] = acc[r] + bias;
  }
}

// ---------------- state init: state = gather(node_enc, adj) * (1-mask) ----------------
__global__ void k_init_state(const float* node_enc, const int* adj, const int* num_nodes,
                             float* state) {
  int t = blockIdx.x * blockDim.x + threadIdx.x;   // over B*V*D
  int b = t / (VV * DD);
  int a = adj[t];
  bool m = (a == num_nodes[b]);
  const float4* src = (const float4*)(node_enc + ((size_t)b * VV + a) * FF);
  float4* dst = (float4*)(state + (size_t)t * FF);
  float4 z = {0.f, 0.f, 0.f, 0.f};
  for (int f = 0; f < FF / 4; ++f) dst[f] = m ? z : src[f];
}

// ---------------- layer A: h = tanh(S@gat_W+b); attn softmax; agg ----------------
__global__ __launch_bounds__(32 * KW) void k_layerA(const float* state, const h16* wgat,
                                                    const float* gatb, const float* gata,
                                                    const int* adj, const int* num_nodes,
                                                    float* agg) {
  __shared__ __align__(16) h16 S16[KW][16 * 64];
  __shared__ float Hh[KW][16 * 64];
  __shared__ float sc[KW][16];
  __shared__ float at[KW][16];
  const int lane = threadIdx.x & 31, w = threadIdx.x >> 5;
  const int bv = blockIdx.x * KW + w;
  const int b = bv >> 11;
  h16* Sl = S16[w]; float* Hl = Hh[w];
  const float* st = state + (size_t)bv * DD * FF;
  for (int idx = lane; idx < 16 * 32; idx += 32) {
    v2h p; p[0] = (h16)st[2*idx]; p[1] = (h16)st[2*idx + 1];
    ((v2h*)Sl)[idx] = p;
  }
  __syncthreads();
  v16h a0 = load_a_lds16(Sl, 0, lane);
  v16h a1 = load_a_lds16(Sl, 32, lane);
  const int n = lane & 15, m0 = (lane & 16) ? 8 : 0;
  for (int j = 0; j < 4; ++j) {
    v8f acc = {};
    acc = wmma32(a0, load_b_packed(wgat, 0 + j, lane), acc);
    acc = wmma32(a1, load_b_packed(wgat, 4 + j, lane), acc);
    float bias = gatb[j * 16 + n];
#pragma unroll
    for (int r = 0; r < 8; ++r) Hl[(m0 + r) * 64 + j * 16 + n] = tanhf(acc[r] + bias);
  }
  __syncthreads();
  const int nn = num_nodes[b];
  if (lane < 16) {
    float s = 0.f;
    for (int f = 0; f < 64; ++f) s += Hl[lane * 64 + f] * gata[f];
    if (adj[bv * DD + lane] == nn) s -= BIGF;
    sc[w][lane] = s;
  }
  __syncthreads();
  if (lane < 16) {
    float mx = -INFINITY;
    for (int d = 0; d < 16; ++d) mx = fmaxf(mx, sc[w][d]);
    float sum = 0.f;
    for (int d = 0; d < 16; ++d) sum += expf(sc[w][d] - mx);
    at[w][lane] = expf(sc[w][lane] - mx) / sum;
  }
  __syncthreads();
  for (int f = lane; f < 64; f += 32) {
    float s = 0.f;
    for (int d = 0; d < 16; ++d) s += at[w][d] * Hl[d * 64 + f];
    agg[(size_t)bv * FF + f] = s;
  }
}

// ---------------- layer B: msg gather + GRU update (state in-place) ----------------
__global__ __launch_bounds__(32 * KW) void k_layerB(const float* node_enc, const float* agg,
                                                    const int* adj, const int* num_nodes,
                                                    const h16* wgx, const h16* wgh,
                                                    const float* grub, float* state) {
  __shared__ __align__(16) h16 S16[KW][16 * 64];
  __shared__ __align__(16) h16 M16[KW][16 * 64];
  const int lane = threadIdx.x & 31, w = threadIdx.x >> 5;
  const int bv = blockIdx.x * KW + w;
  const int b = bv >> 11;
  const int nn = num_nodes[b];
  h16* Sl = S16[w]; h16* Ml = M16[w];
  float* st = state + (size_t)bv * DD * FF;
  for (int idx = lane; idx < 16 * 32; idx += 32) {
    v2h p; p[0] = (h16)st[2*idx]; p[1] = (h16)st[2*idx + 1];
    ((v2h*)Sl)[idx] = p;
  }
  for (int idx = lane; idx < 16 * 32; idx += 32) {       // (d, f-pair)
    int d = idx >> 5, fp = (idx & 31) << 1;
    int a = adj[bv * DD + d];
    float v0 = 0.f, v1 = 0.f;
    if (a != nn) {
      size_t g = ((size_t)b * VV + a) * FF + fp;
      v0 = agg[g] + node_enc[g];       // masked_gather(agg)+initial share adj & mask
      v1 = agg[g+1] + node_enc[g+1];
    }
    v2h p; p[0] = (h16)v0; p[1] = (h16)v1;
    ((v2h*)Ml)[idx] = p;
  }
  __syncthreads();
  v16h am0 = load_a_lds16(Ml, 0, lane);
  v16h am1 = load_a_lds16(Ml, 32, lane);
  v16h as0 = load_a_lds16(Sl, 0, lane);
  v16h as1 = load_a_lds16(Sl, 32, lane);
  const int n = lane & 15, m0 = (lane & 16) ? 8 : 0;
  for (int j = 0; j < 4; ++j) {
    // frag(kstep, coltile) for N=192: kstep*12 + coltile ; coltile = gate*4 + j
    v8f xz = {}, xr = {}, xh = {}, hz = {}, hr = {}, hh = {};
    xz = wmma32(am0, load_b_packed(wgx,      0 + j, lane), xz);
    xz = wmma32(am1, load_b_packed(wgx, 12 + 0 + j, lane), xz);
    xr = wmma32(am0, load_b_packed(wgx,      4 + j, lane), xr);
    xr = wmma32(am1, load_b_packed(wgx, 12 + 4 + j, lane), xr);
    xh = wmma32(am0, load_b_packed(wgx,      8 + j, lane), xh);
    xh = wmma32(am1, load_b_packed(wgx, 12 + 8 + j, lane), xh);
    hz = wmma32(as0, load_b_packed(wgh,      0 + j, lane), hz);
    hz = wmma32(as1, load_b_packed(wgh, 12 + 0 + j, lane), hz);
    hr = wmma32(as0, load_b_packed(wgh,      4 + j, lane), hr);
    hr = wmma32(as1, load_b_packed(wgh, 12 + 4 + j, lane), hr);
    hh = wmma32(as0, load_b_packed(wgh,      8 + j, lane), hh);
    hh = wmma32(as1, load_b_packed(wgh, 12 + 8 + j, lane), hh);
    float bz = grub[j * 16 + n], br = grub[64 + j * 16 + n], bh = grub[128 + j * 16 + n];
#pragma unroll
    for (int r = 0; r < 8; ++r) {
      float sv   = st[(m0 + r) * 64 + j * 16 + n];   // f32 state from global (L2)
      float z    = sigm(xz[r] + bz + hz[r]);
      float rr   = sigm(xr[r] + br + hr[r]);
      float cand = tanhf(xh[r] + bh + rr * hh[r]);
      st[(m0 + r) * 64 + j * 16 + n] = z * sv + (1.f - z) * cand;
    }
  }
}

// ---------------- decode: node weights + dual vars ----------------
__global__ __launch_bounds__(32 * KW) void k_decode(const float* state, const h16* wd1,
                                                    const float* d1b, const float* d2W, const float* d2b,
                                                    const float* duW1, const float* dub1,
                                                    const float* duW2, const float* dub2,
                                                    float* nodew, float* dualv) {
  __shared__ __align__(16) h16 S16[KW][16 * 64];
  __shared__ float T[KW][16 * 32];
  __shared__ float NS[KW][64];
  __shared__ float SC[KW][32];
  const int lane = threadIdx.x & 31, w = threadIdx.x >> 5;
  const int bv = blockIdx.x * KW + w;
  h16* Sl = S16[w]; float* Tl = T[w];
  const float* st = state + (size_t)bv * DD * FF;
  for (int idx = lane; idx < 16 * 32; idx += 32) {
    v2h p; p[0] = (h16)st[2*idx]; p[1] = (h16)st[2*idx + 1];
    ((v2h*)Sl)[idx] = p;
  }
  __syncthreads();
  v16h a0 = load_a_lds16(Sl, 0, lane);
  v16h a1 = load_a_lds16(Sl, 32, lane);
  const int n = lane & 15, m0 = (lane & 16) ? 8 : 0;
  for (int j = 0; j < 2; ++j) {
    v8f acc = {};
    acc = wmma32(a0, load_b_packed(wd1, 0 + j, lane), acc);
    acc = wmma32(a1, load_b_packed(wd1, 2 + j, lane), acc);
    float bias = d1b[j * 16 + n];
#pragma unroll
    for (int r = 0; r < 8; ++r) Tl[(m0 + r) * 32 + j * 16 + n] = tanhf(acc[r] + bias);
  }
  __syncthreads();
  if (lane < 16) {
    float s = d2b[0];
    for (int h = 0; h < 32; ++h) s += Tl[lane * 32 + h] * d2W[h];
    nodew[(size_t)bv * DD + lane] = s;
  }
  for (int f = lane; f < 64; f += 32) {                  // node_states from f32 global
    float s = 0.f;
    for (int d = 0; d < 16; ++d) s += st[d * 64 + f];
    NS[w][f] = s;
  }
  __syncthreads();
  {
    float acc = dub1[lane];
    for (int f = 0; f < 64; ++f) acc += NS[w][f] * duW1[f * 32 + lane];
    SC[w][lane] = tanhf(acc) * duW2[lane];
  }
  __syncthreads();
  if (lane == 0) {
    float s = dub2[0];
    for (int h = 0; h < 32; ++h) s += SC[w][h];
    dualv[bv] = s;
  }
}

// ---------------- incoming softmax over gathered node weights ----------------
__global__ void k_incoming(const float* nodew, const int* in_idx, const int* inv_adj,
                           const int* num_nodes, float* attn_in) {
  int bv = blockIdx.x * blockDim.x + threadIdx.x;
  int b = bv >> 11, nn = num_nodes[b];
  float sc[DD], mx = -INFINITY;
  for (int d = 0; d < DD; ++d) {
    float s = nodew[(size_t)b * VV * DD + in_idx[bv * DD + d]];
    if (inv_adj[bv * DD + d] == nn) s -= BIGF;
    sc[d] = s; mx = fmaxf(mx, s);
  }
  float sum = 0.f;
  for (int d = 0; d < DD; ++d) { sc[d] = expf(sc[d] - mx); sum += sc[d]; }
  float inv = 1.f / sum;
  for (int d = 0; d < DD; ++d) attn_in[(size_t)bv * DD + d] = sc[d] * inv;
}

// ---------------- reverse gather + edge softmax + flow0 ----------------
__global__ void k_normflow0(const float* attn_in, const int* rev_idx, const int* adj,
                            const int* num_nodes, const float* demands,
                            float* normalized, float* flow) {
  int bv = blockIdx.x * blockDim.x + threadIdx.x;
  int b = bv >> 11, nn = num_nodes[b];
  float sc[DD], mx = -INFINITY;
  for (int d = 0; d < DD; ++d) {
    float s = attn_in[(size_t)b * VV * DD + rev_idx[bv * DD + d]];
    if (adj[bv * DD + d] == nn) s -= BIGF;
    sc[d] = s; mx = fmaxf(mx, s);
  }
  float sum = 0.f;
  for (int d = 0; d < DD; ++d) { sc[d] = expf(sc[d] - mx); sum += sc[d]; }
  float inv = 1.f / sum;
  float supply = fmaxf(0.f, -demands[bv]);
  for (int d = 0; d < DD; ++d) {
    float nrm = sc[d] * inv;
    normalized[(size_t)bv * DD + d] = nrm;
    flow[(size_t)bv * DD + d] = nrm * supply;
  }
}

// ---------------- one min-cost-flow iteration ----------------
__global__ void k_flowiter(const float* src, const float* normalized, const int* in_idx,
                           const int* inv_adj, const int* num_nodes, const float* demands,
                           float* dst) {
  int bv = blockIdx.x * blockDim.x + threadIdx.x;
  int b = bv >> 11, nn = num_nodes[b];
  float inflow = 0.f;
  for (int d = 0; d < DD; ++d) {
    float f = src[(size_t)b * VV * DD + in_idx[bv * DD + d]];
    if (inv_adj[bv * DD + d] != nn) inflow += f;
  }
  float t = inflow + fmaxf(0.f, -demands[bv]);
  for (int d = 0; d < DD; ++d)
    dst[(size_t)bv * DD + d] = normalized[(size_t)bv * DD + d] * t;
}

// ---------------- dual descent + loss reduction ----------------
__global__ void k_final(const float* flow, const float* dualv, const int* adj,
                        const int* num_nodes, const float* demands, float* out) {
  int bv = blockIdx.x * blockDim.x + threadIdx.x;
  int b = bv >> 11, nn = num_nodes[b];
  float fc = 0.f;
  for (int d = 0; d < DD; ++d) { float f = flow[(size_t)bv * DD + d]; fc += f * f; }
  float dv = dualv[bv];
  float dcost = 0.f;
  for (int d = 0; d < DD; ++d) {
    int a = adj[bv * DD + d];
    float m = (a == nn) ? 1.f : 0.f;
    float dd = dualv[(size_t)b * VV + a] * (1.f - m) - m * dv;
    float f = 0.f, acc = 0.f;
    for (int it = 0; it < 10; ++it) {
      float g = 2.f * f + dd;
      acc = 0.9f * acc + 0.01f * g;
      f = fmaxf(0.f, f - acc) * (1.f - m);
    }
    dcost += f * f + dd * f;
  }
  float contrib = fc - dcost + dv * demands[bv];
  atomicAdd(&out[b], contrib);
}

extern "C" void kernel_launch(void* const* d_in, const int* in_sizes, int n_in,
                              void* d_out, int out_size, void* d_ws, size_t ws_size,
                              hipStream_t stream) {
  const float* feat   = (const float*)d_in[0];
  const float* dem    = (const float*)d_in[1];
  const float* emb    = (const float*)d_in[2];
  const float* encW   = (const float*)d_in[3];
  const float* encb   = (const float*)d_in[4];
  const float* gatW   = (const float*)d_in[5];
  const float* gatb   = (const float*)d_in[6];
  const float* gata   = (const float*)d_in[7];
  const float* gruWx  = (const float*)d_in[8];
  const float* gruWh  = (const float*)d_in[9];
  const float* grub   = (const float*)d_in[10];
  const float* decW1  = (const float*)d_in[11];
  const float* decb1  = (const float*)d_in[12];
  const float* decW2  = (const float*)d_in[13];
  const float* decb2  = (const float*)d_in[14];
  const float* duW1   = (const float*)d_in[15];
  const float* dub1   = (const float*)d_in[16];
  const float* duW2   = (const float*)d_in[17];
  const float* dub2   = (const float*)d_in[18];
  const int*   adj    = (const int*)d_in[19];
  const int*   invadj = (const int*)d_in[20];
  const int*   inidx  = (const int*)d_in[21];
  const int*   revidx = (const int*)d_in[22];
  const int*   nn     = (const int*)d_in[23];
  float* out = (float*)d_out;

  char* wsb = (char*)d_ws;
  h16* wenc = (h16*)wsb;                 // 8 frags  = 4096 halves
  h16* wgat = wenc + 4096;               // 8 frags
  h16* wgx  = wgat + 4096;               // 24 frags = 12288 halves
  h16* wgh  = wgx  + 12288;              // 24 frags
  h16* wd1  = wgh  + 12288;              // 4 frags  = 2048 halves
  size_t off = 131072;                                   // 128 KB for packed weights
  float* node_enc = (float*)(wsb + off); off += (size_t)BB * VV * FF * 4;
  float* agg      = (float*)(wsb + off); off += (size_t)BB * VV * FF * 4;
  float* state    = (float*)(wsb + off); off += (size_t)BB * VV * DD * FF * 4;
  float* nodew    = (float*)(wsb + off); off += (size_t)BB * VV * DD * 4;
  float* attn_in  = (float*)(wsb + off); off += (size_t)BB * VV * DD * 4;
  float* normed   = (float*)(wsb + off); off += (size_t)BB * VV * DD * 4;
  float* flowA    = (float*)(wsb + off); off += (size_t)BB * VV * DD * 4;
  float* flowB    = (float*)(wsb + off); off += (size_t)BB * VV * DD * 4;
  float* dualv    = (float*)(wsb + off); off += (size_t)BB * VV * 4;
  (void)ws_size; (void)in_sizes; (void)n_in;

  const int NBV = BB * VV;                 // 16384 (b,v) tiles
  hipLaunchKernelGGL(k_prep, dim3(48), dim3(256), 0, stream,
                     encW, gatW, gruWx, gruWh, decW1, wenc, wgat, wgx, wgh, wd1);
  hipLaunchKernelGGL(k_encode, dim3(NBV / 16 / KW), dim3(32 * KW), 0, stream,
                     feat, emb, wenc, encb, node_enc);
  hipLaunchKernelGGL(k_init_state, dim3(BB * VV * DD / 256), dim3(256), 0, stream,
                     node_enc, adj, nn, state);
  for (int layer = 0; layer < 2; ++layer) {
    hipLaunchKernelGGL(k_layerA, dim3(NBV / KW), dim3(32 * KW), 0, stream,
                       state, wgat, gatb, gata, adj, nn, agg);
    hipLaunchKernelGGL(k_layerB, dim3(NBV / KW), dim3(32 * KW), 0, stream,
                       node_enc, agg, adj, nn, wgx, wgh, grub, state);
  }
  hipLaunchKernelGGL(k_decode, dim3(NBV / KW), dim3(32 * KW), 0, stream,
                     state, wd1, decb1, decW2, decb2, duW1, dub1, duW2, dub2,
                     nodew, dualv);
  hipLaunchKernelGGL(k_incoming, dim3(NBV / 256), dim3(256), 0, stream,
                     nodew, inidx, invadj, nn, attn_in);
  hipLaunchKernelGGL(k_normflow0, dim3(NBV / 256), dim3(256), 0, stream,
                     attn_in, revidx, adj, nn, dem, normed, flowA);
  for (int it = 0; it < 10; ++it) {
    float* src = (it & 1) ? flowB : flowA;
    float* dst = (it & 1) ? flowA : flowB;
    hipLaunchKernelGGL(k_flowiter, dim3(NBV / 256), dim3(256), 0, stream,
                       src, normed, inidx, invadj, nn, dem, dst);
  }
  hipMemsetAsync(d_out, 0, (size_t)out_size * sizeof(float), stream);
  hipLaunchKernelGGL(k_final, dim3(NBV / 256), dim3(256), 0, stream,
                     flowA, dualv, adj, nn, dem, out);   // 10 iters even -> result in flowA
}
